// percnn_time_stepping_56315611185389
// MI455X (gfx1250) — compile-verified, matrix operand order checked by reading, not run on hardware
//
#include <hip/hip_runtime.h>

// ---------------------------------------------------------------------------
// PerCNN explicit time step on MI455X (gfx1250, wave32).
//
// Roofline: 8.39M pixels; 8B read (u,v of last slice) + 8B written per pixel
// = 134 MB => ~5.8 us at 23.3 TB/s.  Pure per-lane VALU needs ~360 wave-insts
// per 32 pixels (several x the memory time), so the 1x1 convs run on WMMA:
//   channels(16)=M, pixels(16)=N, cin(+bias-row)=K
//   * hidden convs : V_WMMA_F32_16X16X4_F32 (exact f32; bias folded via a
//     1.0 K-row; unused K slots zeroed in A so garbage B lanes contribute 0)
//   * Hadamard prods: elementwise f32 (compiler emits v_pk_mul_f32)
//   * output dots  : V_WMMA_F32_16X16X32_F16 with K fully packed:
//       - per sub-tile: row0 = w4u . pu (K0-15), row1 = w4v . pv (K16-31)
//       - one shared WMMA: rows0-3 = w4s{0,1} . ps_st{0,1}
//     f16 only touches the DT-scaled correction (~1e-6 relative on output).
// Each wave runs two 16-pixel sub-tiles per iteration (full 128B/wave memory
// ops); lane halves are exchanged with ds_swizzle SWAPX16.  Uniform scalar
// trip count (16) keeps EXEC all-ones (WMMA requirement) and loop control in
// SALU only; prefetch address wraps mod NPIX so it needs no guard.
// ---------------------------------------------------------------------------

typedef float    v2f    __attribute__((ext_vector_type(2)));
typedef float    v8f    __attribute__((ext_vector_type(8)));
typedef _Float16 v16h   __attribute__((ext_vector_type(16)));
typedef _Float16 half2t __attribute__((ext_vector_type(2)));

#define C_DU 0.001f
#define C_DV 0.005f
#define C_DT 0.05f

static constexpr int    HW     = 512 * 512;            // 1 << 18
static constexpr size_t NPIX   = (size_t)32 * HW;      // 2^23 = 8,388,608
static constexpr int    BLOCKX = 256;
static constexpr int    GRIDX  = 2048;
static constexpr size_t STRIDE = (size_t)BLOCKX * GRIDX;  // 524288
static constexpr int    ITERS  = (int)(NPIX / STRIDE);    // 16 exactly

union V16U { v16h v; half2t h[8]; };

__device__ __forceinline__ half2t pk2(float a, float b) {
  return __builtin_bit_cast(half2t, __builtin_amdgcn_cvt_pkrtz(a, b));
}
__device__ __forceinline__ float swapx16(float x) {
  // group-of-32 swizzle: and=0x1F, or=0, xor=0x10 -> lane ^= 16
  return __builtin_bit_cast(float,
      __builtin_amdgcn_ds_swizzle(__builtin_bit_cast(int, x), 0x401F));
}
__device__ __forceinline__ v8f wmma_k16(v16h a, v16h b, v8f c) {
  return __builtin_amdgcn_wmma_f32_16x16x32_f16(false, a, false, b, (short)0,
                                                c, false, false);
}
__device__ __forceinline__ v8f wmma_k4(v2f a, v2f b, v8f c) {
  return __builtin_amdgcn_wmma_f32_16x16x4_f32(false, a, false, b, (short)0,
                                               c, false, false);
}

// Build an f16 16x32 A matrix from a weight functor W(m, k) (m = row,
// k = 0..31).  Layout: VGPR r, lanes0-15 hold K=(kl,kl+1), lanes16-31 hold
// K=(kl+8,kl+9), where kl = 2r (r<4) or 2r+8 (r>=4).
template <typename F>
__device__ __forceinline__ v16h build_final_A(int lane, F W) {
  V16U A = {};
  const int m   = lane & 15;
  const int add = (lane & 16) ? 8 : 0;
#pragma unroll
  for (int r = 0; r < 8; ++r) {
    const int k = 2 * r + (r >= 4 ? 8 : 0) + add;
    A.h[r] = pk2(W(m, k), W(m, k + 1));
  }
  return A.v;
}

__global__ void __launch_bounds__(BLOCKX)
percnn_step(const float* __restrict__ x, float* __restrict__ out,
            const float* __restrict__ w1u, const float* __restrict__ b1u,
            const float* __restrict__ w2u, const float* __restrict__ b2u,
            const float* __restrict__ w3u, const float* __restrict__ b3u,
            const float* __restrict__ w4u, const float* __restrict__ b4u,
            const float* __restrict__ w1v, const float* __restrict__ b1v,
            const float* __restrict__ w2v, const float* __restrict__ b2v,
            const float* __restrict__ w3v, const float* __restrict__ b3v,
            const float* __restrict__ w4v, const float* __restrict__ b4v,
            const float* __restrict__ w1s, const float* __restrict__ b1s,
            const float* __restrict__ w2s, const float* __restrict__ b2s,
            const float* __restrict__ w3s, const float* __restrict__ b3s,
            const float* __restrict__ w4s, const float* __restrict__ b4s) {
  const int  lane = (int)(threadIdx.x & 31u);
  const int  mrow = lane & 15;
  const bool lo   = lane < 16;

  const float* wu[3] = {w1u, w2u, w3u};
  const float* bu[3] = {b1u, b2u, b3u};
  const float* wv[3] = {w1v, w2v, w3v};
  const float* bv[3] = {b1v, b2v, b3v};
  const float* ws[3] = {w1s, w2s, w3s};
  const float* bs[3] = {b1s, b2s, b3s};

  // ----- hidden-layer A matrices (f32 16x4, wave-uniform, built once) -----
  // A layout: VGPR0 = K0 (lanes0-15) / K2 (lanes16-31), VGPR1 = K1 / K3.
  v2f Au[3], Av[3], As[3];
#pragma unroll
  for (int j = 0; j < 3; ++j) {
    Au[j].x = lo ? wu[j][mrow] : 0.0f;            // K0 = weight, K2 = 0
    Au[j].y = lo ? bu[j][mrow] : 0.0f;            // K1 = bias,   K3 = 0
    Av[j].x = lo ? wv[j][mrow] : 0.0f;
    Av[j].y = lo ? bv[j][mrow] : 0.0f;
    As[j].x = lo ? ws[j][2 * mrow + 0] : bs[j][mrow];  // K0 = w(:,0), K2 = b
    As[j].y = lo ? ws[j][2 * mrow + 1] : 0.0f;         // K1 = w(:,1), K3 = 0
  }

  // ----- final-layer A matrices (f16 16x32, K fully packed) -----
  // uv final (shared by both sub-tiles): row0 = [w4u | 0], row1 = [0 | w4v]
  const v16h AFuv = build_final_A(lane, [&](int m, int k) -> float {
    if (m == 0 && k < 16) return w4u[k];
    if (m == 1 && k >= 16) return w4v[k - 16];
    return 0.0f;
  });
  // s final (both sub-tiles in one WMMA):
  // row0=[w4s0|0] row1=[w4s1|0] row2=[0|w4s0] row3=[0|w4s1]
  const v16h AFs = build_final_A(lane, [&](int m, int k) -> float {
    if (k < 16) {
      if (m == 0) return w4s[k];
      if (m == 1) return w4s[16 + k];
    } else {
      if (m == 2) return w4s[k - 16];
      if (m == 3) return w4s[k];  // 16 + (k - 16)
    }
    return 0.0f;
  });

  const float Ku  = C_DT * (C_DU * b4u[0] + b4s[0]);
  const float Kv  = C_DT * (C_DV * b4v[0] + b4s[1]);
  const float cDU = C_DT * C_DU, cDV = C_DT * C_DV;

  const v8f zc = {};

  size_t p = (size_t)blockIdx.x * BLOCKX + threadIdx.x;
#pragma unroll 1
  for (int it = 0; it < ITERS; ++it, p += STRIDE) {
    const size_t bat = p >> 18;               // batch index
    const size_t i   = p & (size_t)(HW - 1);  // pixel within plane
    // x is (B, T=2, C=2, H, W); uv = x[:,1] -> plane base (4b+2)
    const float* pin = x + ((bat * 4 + 2) << 18) + i;
    const float  uL  = __builtin_nontemporal_load(pin);
    const float  vL  = __builtin_nontemporal_load(pin + HW);

    {  // unconditional prefetch of the next iteration (address wraps mod NPIX)
      const size_t pn = (p + STRIDE) & (NPIX - 1);
      const float* pf =
          x + (((pn >> 18) * 4 + 2) << 18) + (pn & (size_t)(HW - 1));
      __builtin_prefetch(pf, 0, 1);
      __builtin_prefetch(pf + HW, 0, 1);
    }

    const float uS = swapx16(uL);  // pixels 16-31 into lanes 0-15
    const float vS = swapx16(vL);

    V16U  Buv[2];  // per sub-tile final B: K0-15 = pu, K16-31 = pv
    V16U  Bss;     // shared source B:      K0-15 = ps_st0, K16-31 = ps_st1
    float uu[2], vv[2];
#pragma unroll
    for (int st = 0; st < 2; ++st) {
      const float ub = st ? uS : uL;
      const float vb = st ? vS : vL;
      uu[st] = ub;
      vv[st] = vb;

      // f32 4x16 B layout: VGPR0 = row K0 (lanes0-15) / K2 (lanes16-31)
      v2f Bu, Bv, Bs;
      Bu.x = ub;             Bu.y = 1.0f;  // K0=u, K1=1 (bias row)
      Bv.x = vb;             Bv.y = 1.0f;
      Bs.x = lo ? ub : 1.0f; Bs.y = vb;    // K0=u, K2=1, K1=v

      v8f hu[3], hv[3], hs[3];
#pragma unroll
      for (int j = 0; j < 3; ++j) {
        hu[j] = wmma_k4(Au[j], Bu, zc);
        hv[j] = wmma_k4(Av[j], Bv, zc);
        hs[j] = wmma_k4(As[j], Bs, zc);
      }

      // Hadamard products, f32, directly in the C/D register layout
      const v8f pu = hu[0] * hu[1] * hu[2];
      const v8f pv = hv[0] * hv[1] * hv[2];
      const v8f ps = hs[0] * hs[1] * hs[2];

      // Pack prod rows pairwise -> f16 32x16 B striping, all K slots used
#pragma unroll
      for (int r = 0; r < 4; ++r) {
        Buv[st].h[r]     = pk2(pu[2 * r], pu[2 * r + 1]);  // K = 0..15
        Buv[st].h[4 + r] = pk2(pv[2 * r], pv[2 * r + 1]);  // K = 16..31
        Bss.h[st * 4 + r] = pk2(ps[2 * r], ps[2 * r + 1]);
      }
    }

    // 3 fully-packed final WMMAs for all eight 16-deep dots
    const v8f Duv0 = wmma_k16(AFuv, Buv[0].v, zc);
    const v8f Duv1 = wmma_k16(AFuv, Buv[1].v, zc);
    const v8f Dss  = wmma_k16(AFs, Bss.v, zc);

    // out = uv + (DU/DV * (dot + b4) + (dot_s + b4s)) * DT
    const float oU0 = uu[0] + (cDU * Duv0[0] + C_DT * Dss[0] + Ku);
    const float oU1 = uu[1] + (cDU * Duv1[0] + C_DT * Dss[2] + Ku);
    const float oV0 = vv[0] + (cDV * Duv0[1] + C_DT * Dss[1] + Kv);
    const float oV1 = vv[1] + (cDV * Duv1[1] + C_DT * Dss[3] + Kv);

    const float oU1s = swapx16(oU1);  // sub-tile 1 results -> lanes 16-31
    const float oV1s = swapx16(oV1);
    const float outU = lo ? oU0 : oU1s;
    const float outV = lo ? oV0 : oV1s;

    float* po = out + ((bat * 2) << 18) + i;  // out is (B, 2, H, W)
    __builtin_nontemporal_store(outU, po);
    __builtin_nontemporal_store(outV, po + HW);
  }
}

extern "C" void kernel_launch(void* const* d_in, const int* in_sizes, int n_in,
                              void* d_out, int out_size, void* d_ws,
                              size_t ws_size, hipStream_t stream) {
  (void)in_sizes; (void)n_in; (void)out_size; (void)d_ws; (void)ws_size;
  const float* x   = (const float*)d_in[0];
  const float *w1u = (const float*)d_in[1],  *b1u = (const float*)d_in[2],
              *w2u = (const float*)d_in[3],  *b2u = (const float*)d_in[4],
              *w3u = (const float*)d_in[5],  *b3u = (const float*)d_in[6],
              *w4u = (const float*)d_in[7],  *b4u = (const float*)d_in[8],
              *w1v = (const float*)d_in[9],  *b1v = (const float*)d_in[10],
              *w2v = (const float*)d_in[11], *b2v = (const float*)d_in[12],
              *w3v = (const float*)d_in[13], *b3v = (const float*)d_in[14],
              *w4v = (const float*)d_in[15], *b4v = (const float*)d_in[16],
              *w1s = (const float*)d_in[17], *b1s = (const float*)d_in[18],
              *w2s = (const float*)d_in[19], *b2s = (const float*)d_in[20],
              *w3s = (const float*)d_in[21], *b3s = (const float*)d_in[22],
              *w4s = (const float*)d_in[23], *b4s = (const float*)d_in[24];

  // GRIDX*BLOCKX = 524288 threads; NPIX / STRIDE = 16 exactly, so every wave
  // is always full (EXEC all-ones for WMMA) and setup amortizes over 16 iters.
  percnn_step<<<dim3(GRIDX), dim3(BLOCKX), 0, stream>>>(
      x, (float*)d_out,
      w1u, b1u, w2u, b2u, w3u, b3u, w4u, b4u,
      w1v, b1v, w2v, b2v, w3v, b3v, w4v, b4v,
      w1s, b1s, w2s, b2s, w3s, b3s, w4s, b4s);
}
